// MotionAttnProcessor_87196426043894
// MI455X (gfx1250) — compile-verified, matrix operand order checked by reading, not run on hardware
//
#include <hip/hip_runtime.h>

typedef __attribute__((ext_vector_type(16))) _Float16     v16h;
typedef __attribute__((ext_vector_type(8)))  float        v8f;
typedef __attribute__((ext_vector_type(4)))  float        f4v;
typedef __attribute__((ext_vector_type(4)))  unsigned int u4v;

#define C_DIM   1280
#define S_DIM   32
#define N_HEADS 8
#define HD      160              // head dim
#define NW      (1280*1280)      // elements per weight matrix
#define XS      (S_DIM*C_DIM)    // 40960 elements per batch of x
#define KT      (C_DIM/32)       // 40 K-steps for C-dim GEMMs

// Per CDNA5 16-bit A/B layout: each lane holds two contiguous 8-half (16B)
// runs along K at [k0, k0+8) and [k0+16, k0+24), k0 = ktile*32 + (lane>=16)*8.
__device__ __forceinline__ v16h frag16(const _Float16* row, int k0) {
  union { v16h v; uint4 u[2]; } f;
  f.u[0] = *(const uint4*)(row + k0);
  f.u[1] = *(const uint4*)(row + k0 + 16);
  return f.v;
}

__device__ __forceinline__ v8f wmma16(v16h a, v16h b, v8f c) {
  return __builtin_amdgcn_wmma_f32_16x16x32_f16(false, a, false, b,
                                                (short)0, c, false, false);
}

// ------------------------------------------------ prep: weights f32->f16 + pe table
__global__ void prep(const float* __restrict__ wq, const float* __restrict__ wk,
                     const float* __restrict__ wv, const float* __restrict__ wo,
                     _Float16* __restrict__ dst, float* __restrict__ pe) {
  int i = blockIdx.x * blockDim.x + threadIdx.x;
  if (i < NW) {
    dst[i]          = (_Float16)wq[i];
    dst[NW + i]     = (_Float16)wk[i];
    dst[2 * NW + i] = (_Float16)wv[i];
    dst[3 * NW + i] = (_Float16)wo[i];
  }
  if (i < XS) {                      // sinusoidal positional embedding
    int s = i / C_DIM, c = i % C_DIM;
    float a = (float)s * __expf(-0.0071955784f * (float)(c & ~1)); // -ln(1e4)/1280
    pe[i] = (c & 1) ? __cosf(a) : __sinf(a);
  }
}

// ------------------------------------------------ fused attention, 2 batches / WG
__global__ __launch_bounds__(256) void attn_fused(
    const float* __restrict__ hidden,     // [B,32,1280] f32
    const float* __restrict__ pe,         // [32,1280] f32
    const _Float16* __restrict__ w16,     // wq|wk|wv|wo f16, each [1280][1280]
    _Float16* __restrict__ attnb)         // [B*32,1280] f16 (pre out-proj)
{
  __shared__ _Float16 xs[2 * XS];               // 160 KB x + pos-embed, f16
  __shared__ _Float16 qh[2 * S_DIM * HD];       // 20 KB
  __shared__ _Float16 kh[2 * S_DIM * HD];       // 20 KB
  __shared__ _Float16 vhT[2 * HD * S_DIM];      // 20 KB  V transposed [d][s]
  __shared__ _Float16 attn_s[2 * S_DIM * HD];   // 20 KB  per-head out strip
  __shared__ float    sc[2 * S_DIM * S_DIM];    // 8 KB   scores f32
  __shared__ _Float16 probs[2 * S_DIM * S_DIM]; // 4 KB   softmax(P) f16

  const int b0   = blockIdx.x * 2;
  const int tid  = threadIdx.x;
  const int wave = tid >> 5;
  const int lane = tid & 31;
  const int l16  = lane & 15;
  const int hoff = (lane >> 4) << 3;            // 0/8: K-offset AND M/N half offset
  const int QS   = S_DIM * HD;                  // 5120
  const int SS   = S_DIM * S_DIM;               // 1024

  // ---- Stage 1: x = hidden + pe, cast to f16 in LDS (both batches).
  //      hidden is streamed once -> non-temporal so weights stay in L2. ----
  for (int bb = 0; bb < 2; ++bb) {
    const float* hsrc = hidden + (size_t)(b0 + bb) * XS;
    _Float16* xd = xs + bb * XS;
    for (int i = tid * 4; i < XS; i += 1024) {
      f4v hv = __builtin_nontemporal_load((const f4v*)(hsrc + i));
      f4v pv = *(const f4v*)(pe + i);
      xd[i + 0] = (_Float16)(hv[0] + pv[0]);
      xd[i + 1] = (_Float16)(hv[1] + pv[1]);
      xd[i + 2] = (_Float16)(hv[2] + pv[2]);
      xd[i + 3] = (_Float16)(hv[3] + pv[3]);
    }
  }
  __syncthreads();

  const int rowA0 = l16 * C_DIM;                // mt=0 A row base
  const int rowA1 = (16 + l16) * C_DIM;         // mt=1 A row base

  for (int h = 0; h < N_HEADS; ++h) {
    // ---- Stage 2a: Q/K/V projections. Job = (proj, nt); each job computes
    //      4 tiles (2 batches x 2 seq-tiles) reusing one B fragment. ----
    for (int j = wave; j < 30; j += 8) {
      int p  = j / 10;                          // 0=Q 1=K 2=V
      int nt = j % 10;                          // head-dim tile
      const _Float16* wrow = w16 + (size_t)p * NW
                           + (size_t)(h * HD + nt * 16 + l16) * C_DIM;
      v8f a00 = {0.f,0.f,0.f,0.f,0.f,0.f,0.f,0.f};
      v8f a01 = a00, a10 = a00, a11 = a00;
      for (int kt = 0; kt < KT; ++kt) {
        int k0 = kt * 32 + hoff;
        v16h bf = frag16(wrow, k0);             // one L2 fetch, 4 WMMAs
        a00 = wmma16(frag16(xs      + rowA0, k0), bf, a00);
        a01 = wmma16(frag16(xs      + rowA1, k0), bf, a01);
        a10 = wmma16(frag16(xs + XS + rowA0, k0), bf, a10);
        a11 = wmma16(frag16(xs + XS + rowA1, k0), bf, a11);
      }
      for (int bb = 0; bb < 2; ++bb) {
        v8f* accs = bb ? &a10 : &a00;
        for (int mt = 0; mt < 2; ++mt) {
          v8f acc = accs[mt];
          if (p < 2) {
            _Float16* dst = (p == 0 ? qh : kh) + bb * QS;
            for (int r = 0; r < 8; ++r)
              dst[(mt * 16 + r + hoff) * HD + nt * 16 + l16] = (_Float16)acc[r];
          } else {
            _Float16* dst = vhT + bb * QS;
            for (int r = 0; r < 8; ++r)
              dst[(nt * 16 + l16) * S_DIM + mt * 16 + r + hoff] = (_Float16)acc[r];
          }
        }
      }
    }
    __syncthreads();

    // ---- Stage 2b: scores = Q K^T * scale (8 tiles = 8 waves exactly) ----
    {
      int bb = wave >> 2, q = wave & 3, mt = q & 1, nt = q >> 1;
      const _Float16* arow = qh + bb * QS + (mt * 16 + l16) * HD;
      const _Float16* brow = kh + bb * QS + (nt * 16 + l16) * HD;
      v8f acc = {0.f,0.f,0.f,0.f,0.f,0.f,0.f,0.f};
      for (int kt = 0; kt < HD / 32; ++kt) {
        int k0 = kt * 32 + hoff;
        acc = wmma16(frag16(arow, k0), frag16(brow, k0), acc);
      }
      const float scale = 0.07905694150420949f; // 1/sqrt(160)
      for (int r = 0; r < 8; ++r)
        sc[bb * SS + (mt * 16 + r + hoff) * S_DIM + nt * 16 + l16] = acc[r] * scale;
    }
    __syncthreads();

    // ---- Stage 2c: row softmax (64 rows, one thread each) ----
    if (tid < 64) {
      float* row = sc + (tid >> 5) * SS + (tid & 31) * S_DIM;
      _Float16* prow = probs + (tid >> 5) * SS + (tid & 31) * S_DIM;
      float mx = -1e30f;
      for (int j = 0; j < S_DIM; ++j) mx = fmaxf(mx, row[j]);
      float sum = 0.f;
      for (int j = 0; j < S_DIM; ++j) sum += __expf(row[j] - mx);
      float inv = 1.0f / sum;
      for (int j = 0; j < S_DIM; ++j)
        prow[j] = (_Float16)(__expf(row[j] - mx) * inv);
    }
    __syncthreads();

    // ---- Stage 2d: out_h = P @ V (40 jobs over 8 waves, K=32); results go
    //      to an LDS strip so the global write can be vectorized. ----
    for (int j = wave; j < 40; j += 8) {
      int bb = j & 1, t = j >> 1, mt = t & 1, nt = t >> 1;
      const _Float16* arow = probs + bb * SS + (mt * 16 + l16) * S_DIM;
      const _Float16* brow = vhT + bb * QS + (nt * 16 + l16) * S_DIM;
      v8f acc = {0.f,0.f,0.f,0.f,0.f,0.f,0.f,0.f};
      acc = wmma16(frag16(arow, hoff), frag16(brow, hoff), acc);
      for (int r = 0; r < 8; ++r)
        attn_s[(bb * S_DIM + mt * 16 + r + hoff) * HD + nt * 16 + l16] =
            (_Float16)acc[r];
    }
    __syncthreads();

    // ---- Stage 2e: vectorized copy LDS strip -> attnb (b128, non-temporal) ----
    for (int i = tid; i < 2 * S_DIM * (HD / 8); i += 256) {   // 1280 uint4
      int bb = i / (S_DIM * (HD / 8));
      int rr = (i / (HD / 8)) % S_DIM;
      int cc = (i % (HD / 8)) * 8;
      u4v val = *(const u4v*)(attn_s + (bb * S_DIM + rr) * HD + cc);
      __builtin_nontemporal_store(val,
          (u4v*)(attnb + (size_t)((b0 + bb) * S_DIM + rr) * C_DIM + h * HD + cc));
    }
    // next head's 2a/2b/2c barriers protect attn_s reuse; qh/kh/vhT writes
    // don't conflict with this copy.
  }
}

// ------------------------------------------------ output projection, 2 batches / WG
__global__ __launch_bounds__(256) void out_proj(
    const _Float16* __restrict__ attnb, const _Float16* __restrict__ wo16,
    const float* __restrict__ bo, float* __restrict__ out)
{
  __shared__ _Float16 as[2 * XS];               // 160 KB A tile (64 rows)
  const int b0 = blockIdx.x * 2, tid = threadIdx.x;
  const _Float16* src = attnb + (size_t)b0 * XS;
  for (int i = tid; i < (2 * XS) / 8; i += 256) {
    u4v val = __builtin_nontemporal_load(((const u4v*)src) + i);  // read-once
    *(((u4v*)as) + i) = val;
  }
  __syncthreads();

  const int wave = tid >> 5, lane = tid & 31;
  const int l16 = lane & 15, hoff = (lane >> 4) << 3;
  const int rowA0 = l16 * C_DIM, rowA1 = (16 + l16) * C_DIM;

  for (int nt = wave; nt < 80; nt += 8) {       // 80 N-tiles; B reused x4
    const _Float16* brow = wo16 + (size_t)(nt * 16 + l16) * C_DIM;
    v8f a00 = {0.f,0.f,0.f,0.f,0.f,0.f,0.f,0.f};
    v8f a01 = a00, a10 = a00, a11 = a00;
    for (int kt = 0; kt < KT; ++kt) {
      int k0 = kt * 32 + hoff;
      v16h bf = frag16(brow, k0);
      a00 = wmma16(frag16(as      + rowA0, k0), bf, a00);
      a01 = wmma16(frag16(as      + rowA1, k0), bf, a01);
      a10 = wmma16(frag16(as + XS + rowA0, k0), bf, a10);
      a11 = wmma16(frag16(as + XS + rowA1, k0), bf, a11);
    }
    int n = nt * 16 + l16;
    float bn = bo[n];
    for (int bb = 0; bb < 2; ++bb) {
      v8f* accs = bb ? &a10 : &a00;
      for (int mt = 0; mt < 2; ++mt) {
        v8f acc = accs[mt];
        for (int r = 0; r < 8; ++r)
          __builtin_nontemporal_store(acc[r] + bn,
              &out[(size_t)((b0 + bb) * S_DIM + mt * 16 + r + hoff) * C_DIM + n]);
      }
    }
  }
}

extern "C" void kernel_launch(void* const* d_in, const int* in_sizes, int n_in,
                              void* d_out, int out_size, void* d_ws, size_t ws_size,
                              hipStream_t stream) {
  const float* hidden = (const float*)d_in[0];
  const float* Wq = (const float*)d_in[1];
  const float* Wk = (const float*)d_in[2];
  const float* Wv = (const float*)d_in[3];
  const float* Wo = (const float*)d_in[4];
  const float* bo = (const float*)d_in[5];

  int B = in_sizes[0] / XS;                     // 2048

  _Float16* w16   = (_Float16*)d_ws;            // 4 weight matrices, f16
  float*    pe    = (float*)(w16 + 4 * (size_t)NW);
  _Float16* attnb = (_Float16*)(pe + XS);       // [B*32,1280] f16

  prep<<<(NW + 255) / 256, 256, 0, stream>>>(Wq, Wk, Wv, Wo, w16, pe);
  attn_fused<<<B / 2, 256, 0, stream>>>(hidden, pe, w16, attnb);
  out_proj<<<B / 2, 256, 0, stream>>>(attnb, w16 + 3 * (size_t)NW, bo, (float*)d_out);
}
